// RWKV_TimeMix_65798898975556
// MI455X (gfx1250) — matrix operand order repaired and checked by reading, not compiled, not run on hardware
//
#include <hip/hip_runtime.h>
#include <stdint.h>
#include <stddef.h>

// Problem constants (match reference)
#define B_   4
#define T_   2048
#define D_   1024
#define H_   16
#define N_   64
#define BT_  (B_ * T_)            // 8192
#define BTD_ (B_ * T_ * D_)       // 8388608
#define DW_  64
#define DA_  64
#define DV_  32
#define DG_  128

typedef __attribute__((ext_vector_type(16))) __bf16 v16bf;
typedef __attribute__((ext_vector_type(8)))  float  v8f;

struct __align__(16) U4 { unsigned int a, b, c, d; };

__device__ __forceinline__ unsigned short f2bf(float f) {
    unsigned int u = __builtin_bit_cast(unsigned int, f);
    unsigned int r = u + 0x7FFFu + ((u >> 16) & 1u);   // round-to-nearest-even
    return (unsigned short)(r >> 16);
}
__device__ __forceinline__ float sigm(float x) { return 1.0f / (1.0f + __expf(-x)); }
__device__ __forceinline__ float tanh_(float x) { float e = __expf(2.0f * x); return (e - 1.0f) / (e + 1.0f); }

// ---------------------------------------------------------------------------
// CDNA5 async global -> LDS copy (16 bytes / lane), ASYNCcnt-tracked.
// VGLOBAL encoding (ISA 15.18.3 op 98): vdst = LDS byte address, vaddr = 64-bit
// global address, saddr = off. Generic LDS pointers keep the LDS byte offset in
// their low 32 bits (ISA 10.2 aperture mapping), so a truncation suffices.
// ---------------------------------------------------------------------------
__device__ __forceinline__ void async_copy16(const unsigned short* g, unsigned short* l) {
    unsigned int lds_off = (unsigned int)(unsigned long long)(const void*)l;
    asm volatile("global_load_async_to_lds_b128 %0, %1, off"
                 :
                 : "v"(lds_off), "v"(g)
                 : "memory");
}
__device__ __forceinline__ void wait_async() {
#if __has_builtin(__builtin_amdgcn_s_wait_asynccnt)
    __builtin_amdgcn_s_wait_asynccnt(0);
#else
    asm volatile("s_wait_asynccnt 0x0" ::: "memory");
#endif
}

// ---------------------------------------------------------------------------
// 1) fp32 weight [K][N] -> bf16 transposed [N][K] (one-time prep; makes the
//    GEMM B-tile staging identical to the A-tile staging: contiguous rows).
// ---------------------------------------------------------------------------
__global__ void cvt_transpose_bf16(const float* __restrict__ s, unsigned short* __restrict__ d,
                                   int K, int N) {
    int i = blockIdx.x * blockDim.x + threadIdx.x;   // index into dst [N][K]
    if (i >= K * N) return;
    int n = i / K, k = i - n * K;
    d[i] = f2bf(s[(size_t)k * N + n]);
}

// ---------------------------------------------------------------------------
// 2) token shift + six lerps, store bf16 activations
// ---------------------------------------------------------------------------
__global__ __launch_bounds__(256)
void prep_lerp(const float* __restrict__ x, const float* __restrict__ shift_prev,
               const float* __restrict__ mr, const float* __restrict__ mk,
               const float* __restrict__ mv, const float* __restrict__ md,
               const float* __restrict__ ma, const float* __restrict__ mg,
               unsigned short* __restrict__ lr, unsigned short* __restrict__ lk,
               unsigned short* __restrict__ lv, unsigned short* __restrict__ ld,
               unsigned short* __restrict__ la, unsigned short* __restrict__ lg) {
    int i = blockIdx.x * blockDim.x + threadIdx.x;
    if (i >= BTD_) return;
    int d  = i & (D_ - 1);
    int bt = i >> 10;                 // D_ = 1024
    int t  = bt & (T_ - 1);
    int b  = bt >> 11;                // T_ = 2048
    float xc = x[i];
    float xs = (t == 0) ? shift_prev[b * D_ + d] : x[i - D_];
    float dx = xs - xc;
    lr[i] = f2bf(xc + dx * mr[d]);
    lk[i] = f2bf(xc + dx * mk[d]);
    lv[i] = f2bf(xc + dx * mv[d]);
    ld[i] = f2bf(xc + dx * md[d]);
    la[i] = f2bf(xc + dx * ma[d]);
    lg[i] = f2bf(xc + dx * mg[d]);
}

// ---------------------------------------------------------------------------
// 3) bf16 WMMA GEMM: C[M,N] = A[M,K] * W[K,N], weights given TRANSPOSED as
//    WT[N][K]. block = 128 threads = 4 waves, block tile 64x64, wave 32x32,
//    K step 32, v_wmma_f32_16x16x32_bf16, async global->LDS tile staging.
//    ACT: 0 none, 1 tanh, 2 sigmoid (compile-time). OUTF/OUTB select outputs.
// ---------------------------------------------------------------------------
template<int ACT, bool OUTF, bool OUTB>
__global__ __launch_bounds__(128)
void gemm_bf16(const unsigned short* __restrict__ A, const unsigned short* __restrict__ WT,
               const float* __restrict__ bias, float* __restrict__ Cf,
               unsigned short* __restrict__ Cb, int N, int K) {
    __shared__ unsigned short As[64][32];    // [m][k]
    __shared__ unsigned short Bst[64][32];   // [n][k]

    const int tid  = threadIdx.x;
    const int wave = tid >> 5, lane = tid & 31;
    const int wr = wave >> 1, wc = wave & 1;
    const int half = lane >> 4, l15 = lane & 15;
    const int blockM = blockIdx.y * 64;
    const int blockN = blockIdx.x * 64;

    v8f acc[2][2];
    acc[0][0] = (v8f){}; acc[0][1] = (v8f){}; acc[1][0] = (v8f){}; acc[1][1] = (v8f){};

    const int row = tid >> 1, seg = tid & 1;      // 64 rows x 2 x (16 halfwords)
    const bool brow_ok = (blockN + row) < N;      // N is always a multiple of 16

    for (int k0 = 0; k0 < K; k0 += 32) {
        // stage A tile (async DMA into LDS)
        {
            const unsigned short* ga = A + (size_t)(blockM + row) * K + k0 + seg * 16;
            unsigned short* la = &As[row][seg * 16];
            async_copy16(ga, la);
            async_copy16(ga + 8, la + 8);
        }
        // stage B tile from transposed weights (rows are contiguous in K)
        {
            unsigned short* lb = &Bst[row][seg * 16];
            if (brow_ok) {
                const unsigned short* gb = WT + (size_t)(blockN + row) * K + k0 + seg * 16;
                async_copy16(gb, lb);
                async_copy16(gb + 8, lb + 8);
            } else {
                U4 z = {0u, 0u, 0u, 0u};
                *reinterpret_cast<U4*>(lb)     = z;
                *reinterpret_cast<U4*>(lb + 8) = z;
            }
        }
        wait_async();
        __syncthreads();

        if (k0 + 32 < K) {
            __builtin_prefetch(A + (size_t)(blockM + row) * K + k0 + 32, 0, 1);
            if (brow_ok) __builtin_prefetch(WT + (size_t)(blockN + row) * K + k0 + 32, 0, 1);
        }

        // fragments (ISA wave32 16-bit A 16x32 / B 32x16 layouts)
        union Frag { U4 u[2]; v16bf v; };
        Frag a0, a1, b0, b1;
        {
            int r0 = wr * 32 + l15;
            const unsigned short* p0 = &As[r0][half * 8];
            a0.u[0] = *reinterpret_cast<const U4*>(p0);        // K = half*8 .. +7
            a0.u[1] = *reinterpret_cast<const U4*>(p0 + 16);   // K = 16+half*8 .. +7
            const unsigned short* p1 = &As[r0 + 16][half * 8];
            a1.u[0] = *reinterpret_cast<const U4*>(p1);
            a1.u[1] = *reinterpret_cast<const U4*>(p1 + 16);
            int c0 = wc * 32 + l15;
            const unsigned short* q0 = &Bst[c0][half * 8];     // K = half*8 .. +7, 16+half*8 .. +7
            b0.u[0] = *reinterpret_cast<const U4*>(q0);
            b0.u[1] = *reinterpret_cast<const U4*>(q0 + 16);
            const unsigned short* q1 = &Bst[c0 + 16][half * 8];
            b1.u[0] = *reinterpret_cast<const U4*>(q1);
            b1.u[1] = *reinterpret_cast<const U4*>(q1 + 16);
        }

        acc[0][0] = __builtin_amdgcn_wmma_f32_16x16x32_bf16(false, a0.v, false, b0.v, (short)0, acc[0][0], false, false);
        acc[0][1] = __builtin_amdgcn_wmma_f32_16x16x32_bf16(false, a0.v, false, b1.v, (short)0, acc[0][1], false, false);
        acc[1][0] = __builtin_amdgcn_wmma_f32_16x16x32_bf16(false, a1.v, false, b0.v, (short)0, acc[1][0], false, false);
        acc[1][1] = __builtin_amdgcn_wmma_f32_16x16x32_bf16(false, a1.v, false, b1.v, (short)0, acc[1][1], false, false);
        __syncthreads();
    }

    // epilogue (C/D layout: VGPR i -> M = i + half*8, N = lane&15)
    #pragma unroll
    for (int sa = 0; sa < 2; ++sa) {
        #pragma unroll
        for (int sb = 0; sb < 2; ++sb) {
            int col = blockN + wc * 32 + sb * 16 + l15;
            if (col >= N) continue;
            float bv = bias ? bias[col] : 0.0f;
            #pragma unroll
            for (int i = 0; i < 8; ++i) {
                int rw = blockM + wr * 32 + sa * 16 + half * 8 + i;
                float v = acc[sa][sb][i] + bv;
                if (ACT == 1) v = tanh_(v);
                if (ACT == 2) v = sigm(v);
                size_t off = (size_t)rw * N + col;
                if (OUTF) Cf[off] = v;
                if (OUTB) Cb[off] = f2bf(v);
            }
        }
    }
}

// ---------------------------------------------------------------------------
// 4) per-head post-mix: decay, value residual, k_bar, normalized kappa.
//    one wave per (b,t,h) row of 64; lane handles n and n+32.
// ---------------------------------------------------------------------------
__global__ __launch_bounds__(256)
void post_mix(const float* __restrict__ k_f, const float* __restrict__ dec_o,
              const float* __restrict__ vl, const float* __restrict__ vrg_s,
              const float* __restrict__ vpc, const float* __restrict__ a_vec,
              const float* __restrict__ xi, const float* __restrict__ alpha,
              float* __restrict__ w_vec, float* __restrict__ v_vec,
              float* __restrict__ k_bar, float* __restrict__ kh) {
    int row = blockIdx.x * 8 + (threadIdx.x >> 5);   // (b*T+t)*H + h
    if (row >= BT_ * H_) return;
    int lane = threadIdx.x & 31;
    int h  = row & (H_ - 1);
    int bt = row >> 4;
    size_t base = (size_t)bt * D_ + h * N_;
    const float cdec = 0.60653065971f;               // exp(-0.5)

    float kap[2], kv[2];
    int nn[2] = { lane, lane + 32 };
    #pragma unroll
    for (int j = 0; j < 2; ++j) {
        int d = h * N_ + nn[j];
        kv[j]  = k_f[base + nn[j]];
        kap[j] = kv[j] * xi[d];
    }
    float ss = kap[0] * kap[0] + kap[1] * kap[1];
    #pragma unroll
    for (int m = 16; m >= 1; m >>= 1) ss += __shfl_xor(ss, m, 32);
    float inv = 1.0f / fmaxf(sqrtf(ss), 1e-12f);

    #pragma unroll
    for (int j = 0; j < 2; ++j) {
        size_t i = base + nn[j];
        int d = h * N_ + nn[j];
        w_vec[i] = __expf(-cdec * sigm(dec_o[i]));
        float vp = vpc[i];
        v_vec[i] = vp + (vl[i] - vp) * vrg_s[i];
        k_bar[i] = kv[j] * (1.0f + (a_vec[i] - 1.0f) * alpha[d]);
        kh[i]    = kap[j] * inv;
    }
}

// ---------------------------------------------------------------------------
// 5) WKV scan: one block per (b,h). 256 threads: thread = (m = tid&63, q = tid>>6)
//    owns S[m][16q..16q+15] in registers. Also emits GN stats and S_final.
// ---------------------------------------------------------------------------
__global__ __launch_bounds__(256)
void wkv_scan(const float* __restrict__ S0, const float* __restrict__ r_f,
              const float* __restrict__ w_vec, const float* __restrict__ k_bar,
              const float* __restrict__ v_vec, const float* __restrict__ kh,
              const float* __restrict__ a_vec, float* __restrict__ p_prime,
              float* __restrict__ stats, float* __restrict__ S_out) {
    __shared__ float sv[6][N_];     // r, w, kb, v, kh, a
    __shared__ float red[4][N_];
    __shared__ float red2[4][N_];

    const int tid = threadIdx.x;
    const int m = tid & 63, q = tid >> 6;
    const int bh = blockIdx.x, b = bh >> 4, h = bh & 15;

    float S[16];
    {
        size_t sb = ((size_t)bh * N_ + m) * N_ + q * 16;
        #pragma unroll
        for (int j = 0; j < 16; ++j) S[j] = S0[sb + j];
    }
    float gsum = 0.0f, gsq = 0.0f;

    for (int t = 0; t < T_; ++t) {
        size_t vb = ((size_t)(b * T_ + t)) * D_ + h * N_;
        for (int pass = 0; pass < 2; ++pass) {
            int j = tid + pass * 256;
            if (j < 6 * N_) {
                int w = j >> 6, n = j & 63;
                const float* p;
                switch (w) {
                    case 0: p = r_f;   break;
                    case 1: p = w_vec; break;
                    case 2: p = k_bar; break;
                    case 3: p = v_vec; break;
                    case 4: p = kh;    break;
                    default: p = a_vec; break;
                }
                sv[w][n] = p[vb + n];
            }
        }
        __syncthreads();

        float skh_p = 0.0f;
        #pragma unroll
        for (int j = 0; j < 16; ++j) skh_p += S[j] * sv[4][q * 16 + j];
        red[q][m] = skh_p;
        __syncthreads();

        float Skh = red[0][m] + red[1][m] + red[2][m] + red[3][m];
        float vm = sv[3][m];
        float rd_p = 0.0f;
        #pragma unroll
        for (int j = 0; j < 16; ++j) {
            int n = q * 16 + j;
            float s = S[j] * sv[1][n] - Skh * (sv[5][n] * sv[4][n]) + vm * sv[2][n];
            S[j] = s;
            rd_p += sv[0][n] * s;
        }
        red2[q][m] = rd_p;
        __syncthreads();

        if (q == 0) {
            float rv = red2[0][m] + red2[1][m] + red2[2][m] + red2[3][m];
            p_prime[vb + m] = rv;
            gsum += rv; gsq += rv * rv;
        }
    }

    __syncthreads();
    if (q == 0) { red[0][m] = gsum; red[1][m] = gsq; }
    __syncthreads();
    if (tid == 0) {
        float s = 0.0f, s2 = 0.0f;
        for (int j = 0; j < 64; ++j) { s += red[0][j]; s2 += red[1][j]; }
        float cnt = (float)T_ * (float)N_;
        float mean = s / cnt;
        stats[bh * 2]     = mean;
        stats[bh * 2 + 1] = s2 / cnt - mean * mean;
    }
    size_t so = ((size_t)bh * N_ + m) * N_ + q * 16;
    #pragma unroll
    for (int j = 0; j < 16; ++j) S_out[so + j] = S[j];
}

// ---------------------------------------------------------------------------
// 6) GroupNorm + bonus + gate, bf16 for final GEMM. One wave per (b,t,h).
// ---------------------------------------------------------------------------
__global__ __launch_bounds__(256)
void gn_bonus_gate(const float* __restrict__ p_prime, const float* __restrict__ stats,
                   const float* __restrict__ r_f, const float* __restrict__ k_bar,
                   const float* __restrict__ v_vec, const float* __restrict__ g_vec,
                   const float* __restrict__ rho, const float* __restrict__ gamma,
                   const float* __restrict__ beta, unsigned short* __restrict__ preout) {
    int row = blockIdx.x * 8 + (threadIdx.x >> 5);
    if (row >= BT_ * H_) return;
    int lane = threadIdx.x & 31;
    int h  = row & (H_ - 1);
    int bt = row >> 4;
    int b  = bt >> 11;                    // T_ = 2048
    size_t base = (size_t)bt * D_ + h * N_;
    float mean = stats[(b * H_ + h) * 2];
    float var  = stats[(b * H_ + h) * 2 + 1];
    float rstd = rsqrtf(var + 1e-5f);

    int nn[2] = { lane, lane + 32 };
    float sc = 0.0f;
    #pragma unroll
    for (int j = 0; j < 2; ++j) {
        size_t i = base + nn[j];
        sc += r_f[i] * (rho[h * N_ + nn[j]] * k_bar[i]);
    }
    #pragma unroll
    for (int m = 16; m >= 1; m >>= 1) sc += __shfl_xor(sc, m, 32);

    #pragma unroll
    for (int j = 0; j < 2; ++j) {
        size_t i = base + nn[j];
        int d = h * N_ + nn[j];
        float pn = (p_prime[i] - mean) * rstd * gamma[d] + beta[d];
        float outv = g_vec[i] * (pn + sc * v_vec[i]);
        preout[i] = f2bf(outv);
    }
}

// ---------------------------------------------------------------------------
// 7) tail: x[:, -1, :] -> second output slot
// ---------------------------------------------------------------------------
__global__ void tail_copy(const float* __restrict__ x, float* __restrict__ out) {
    int i = blockIdx.x * blockDim.x + threadIdx.x;  // B_*D_
    if (i >= B_ * D_) return;
    int b = i >> 10, d = i & (D_ - 1);
    out[i] = x[((size_t)b * T_ + (T_ - 1)) * D_ + d];
}

// ---------------------------------------------------------------------------
extern "C" void kernel_launch(void* const* d_in, const int* in_sizes, int n_in,
                              void* d_out, int out_size, void* d_ws, size_t ws_size,
                              hipStream_t stream) {
    const float* x      = (const float*)d_in[0];
    const float* vpc    = (const float*)d_in[1];
    const float* shiftp = (const float*)d_in[2];
    const float* S0     = (const float*)d_in[3];
    const float* mu_r = (const float*)d_in[4];
    const float* mu_k = (const float*)d_in[5];
    const float* mu_v = (const float*)d_in[6];
    const float* mu_d = (const float*)d_in[7];
    const float* mu_a = (const float*)d_in[8];
    const float* mu_g = (const float*)d_in[9];
    const float* W_r  = (const float*)d_in[10];
    const float* W_k  = (const float*)d_in[11];
    const float* W_vc = (const float*)d_in[12];
    const float* dec1w = (const float*)d_in[13];
    const float* dec1b = (const float*)d_in[14];
    const float* dec2w = (const float*)d_in[15];
    const float* dec2b = (const float*)d_in[16];
    const float* icl1w = (const float*)d_in[17];
    const float* icl1b = (const float*)d_in[18];
    const float* icl2w = (const float*)d_in[19];
    const float* icl2b = (const float*)d_in[20];
    const float* vrg1w = (const float*)d_in[21];
    const float* vrg1b = (const float*)d_in[22];
    const float* vrg2w = (const float*)d_in[23];
    const float* vrg2b = (const float*)d_in[24];
    const float* gat1w = (const float*)d_in[25];
    const float* gat2w = (const float*)d_in[26];
    const float* xi    = (const float*)d_in[27];
    const float* alpha = (const float*)d_in[28];
    const float* rho   = (const float*)d_in[29];
    const float* W_o   = (const float*)d_in[30];
    const float* gam   = (const float*)d_in[31];
    const float* bet   = (const float*)d_in[32];

    float* out_main  = (float*)d_out;                       // B*T*D
    float* out_xlast = out_main + (size_t)BTD_;             // B*D
    float* out_Sfin  = out_xlast + (size_t)B_ * D_;         // B*H*N*N

    char* ws = (char*)d_ws;
    size_t off = 0;
    auto alloc = [&](size_t bytes) -> void* {
        void* p = (void*)(ws + off);
        off += (bytes + 255) & ~(size_t)255;
        return p;
    };

    // bf16 activation buffers
    unsigned short* lx_r = (unsigned short*)alloc((size_t)BTD_ * 2);
    unsigned short* lx_k = (unsigned short*)alloc((size_t)BTD_ * 2);
    unsigned short* lx_v = (unsigned short*)alloc((size_t)BTD_ * 2);
    unsigned short* lx_d = (unsigned short*)alloc((size_t)BTD_ * 2);
    unsigned short* lx_a = (unsigned short*)alloc((size_t)BTD_ * 2);
    unsigned short* lx_g = (unsigned short*)alloc((size_t)BTD_ * 2);
    // bf16 transposed weights ([N][K])
    unsigned short* wb_r   = (unsigned short*)alloc((size_t)D_ * D_ * 2);
    unsigned short* wb_k   = (unsigned short*)alloc((size_t)D_ * D_ * 2);
    unsigned short* wb_vc  = (unsigned short*)alloc((size_t)D_ * D_ * 2);
    unsigned short* wb_o   = (unsigned short*)alloc((size_t)D_ * D_ * 2);
    unsigned short* wb_d1  = (unsigned short*)alloc((size_t)D_ * DW_ * 2);
    unsigned short* wb_d2  = (unsigned short*)alloc((size_t)DW_ * D_ * 2);
    unsigned short* wb_i1  = (unsigned short*)alloc((size_t)D_ * DA_ * 2);
    unsigned short* wb_i2  = (unsigned short*)alloc((size_t)DA_ * D_ * 2);
    unsigned short* wb_v1  = (unsigned short*)alloc((size_t)D_ * DV_ * 2);
    unsigned short* wb_v2  = (unsigned short*)alloc((size_t)DV_ * D_ * 2);
    unsigned short* wb_g1  = (unsigned short*)alloc((size_t)D_ * DG_ * 2);
    unsigned short* wb_g2  = (unsigned short*)alloc((size_t)DG_ * D_ * 2);
    // fp32 intermediates
    float* r_f   = (float*)alloc((size_t)BTD_ * 4);
    float* k_f   = (float*)alloc((size_t)BTD_ * 4);
    float* vl_f  = (float*)alloc((size_t)BTD_ * 4);
    float* dec_o = (float*)alloc((size_t)BTD_ * 4);
    float* a_vec = (float*)alloc((size_t)BTD_ * 4);
    float* vrg_s = (float*)alloc((size_t)BTD_ * 4);
    float* g_vec = (float*)alloc((size_t)BTD_ * 4);
    float* w_vec = (float*)alloc((size_t)BTD_ * 4);
    float* v_vec = (float*)alloc((size_t)BTD_ * 4);
    float* k_bar = (float*)alloc((size_t)BTD_ * 4);
    float* kh    = (float*)alloc((size_t)BTD_ * 4);
    float* pprm  = (float*)alloc((size_t)BTD_ * 4);
    float* stats = (float*)alloc((size_t)B_ * H_ * 2 * 4);
    // bf16 hidden activations
    unsigned short* dec_h = (unsigned short*)alloc((size_t)BT_ * DW_ * 2);
    unsigned short* icl_h = (unsigned short*)alloc((size_t)BT_ * DA_ * 2);
    unsigned short* vrg_h = (unsigned short*)alloc((size_t)BT_ * DV_ * 2);
    unsigned short* gat_h = (unsigned short*)alloc((size_t)BT_ * DG_ * 2);
    unsigned short* preout = (unsigned short*)alloc((size_t)BTD_ * 2);

    auto cvt = [&](const float* s, unsigned short* d, int K, int N) {
        int n = K * N;
        cvt_transpose_bf16<<<(n + 255) / 256, 256, 0, stream>>>(s, d, K, N);
    };
    cvt(W_r,  wb_r,  D_, D_);   cvt(W_k,  wb_k,  D_, D_);
    cvt(W_vc, wb_vc, D_, D_);   cvt(W_o,  wb_o,  D_, D_);
    cvt(dec1w, wb_d1, D_, DW_); cvt(dec2w, wb_d2, DW_, D_);
    cvt(icl1w, wb_i1, D_, DA_); cvt(icl2w, wb_i2, DA_, D_);
    cvt(vrg1w, wb_v1, D_, DV_); cvt(vrg2w, wb_v2, DV_, D_);
    cvt(gat1w, wb_g1, D_, DG_); cvt(gat2w, wb_g2, DG_, D_);

    prep_lerp<<<BTD_ / 256, 256, 0, stream>>>(x, shiftp, mu_r, mu_k, mu_v, mu_d, mu_a, mu_g,
                                              lx_r, lx_k, lx_v, lx_d, lx_a, lx_g);

    auto grid_for = [&](int N) { return dim3((unsigned)((N + 63) / 64), BT_ / 64, 1); };

    // projections (fp32 out, no activation)
    gemm_bf16<0, true, false><<<grid_for(D_), 128, 0, stream>>>(lx_r, wb_r,  nullptr, r_f,  nullptr, D_, D_);
    gemm_bf16<0, true, false><<<grid_for(D_), 128, 0, stream>>>(lx_k, wb_k,  nullptr, k_f,  nullptr, D_, D_);
    gemm_bf16<0, true, false><<<grid_for(D_), 128, 0, stream>>>(lx_v, wb_vc, nullptr, vl_f, nullptr, D_, D_);
    // MLP first layers (bf16 hidden out)
    gemm_bf16<1, false, true><<<grid_for(DW_), 128, 0, stream>>>(lx_d, wb_d1, dec1b,  nullptr, dec_h, DW_, D_);
    gemm_bf16<0, false, true><<<grid_for(DA_), 128, 0, stream>>>(lx_a, wb_i1, icl1b,  nullptr, icl_h, DA_, D_);
    gemm_bf16<0, false, true><<<grid_for(DV_), 128, 0, stream>>>(lx_v, wb_v1, vrg1b,  nullptr, vrg_h, DV_, D_);
    gemm_bf16<2, false, true><<<grid_for(DG_), 128, 0, stream>>>(lx_g, wb_g1, nullptr, nullptr, gat_h, DG_, D_);
    // MLP second layers (fp32 out)
    gemm_bf16<0, true, false><<<grid_for(D_), 128, 0, stream>>>(dec_h, wb_d2, dec2b,  dec_o, nullptr, D_, DW_);
    gemm_bf16<2, true, false><<<grid_for(D_), 128, 0, stream>>>(icl_h, wb_i2, icl2b,  a_vec, nullptr, D_, DA_);
    gemm_bf16<2, true, false><<<grid_for(D_), 128, 0, stream>>>(vrg_h, wb_v2, vrg2b,  vrg_s, nullptr, D_, DV_);
    gemm_bf16<0, true, false><<<grid_for(D_), 128, 0, stream>>>(gat_h, wb_g2, nullptr, g_vec, nullptr, D_, DG_);

    post_mix<<<(BT_ * H_) / 8, 256, 0, stream>>>(k_f, dec_o, vl_f, vrg_s, vpc, a_vec, xi, alpha,
                                                 w_vec, v_vec, k_bar, kh);

    wkv_scan<<<B_ * H_, 256, 0, stream>>>(S0, r_f, w_vec, k_bar, v_vec, kh, a_vec,
                                          pprm, stats, out_Sfin);

    gn_bonus_gate<<<(BT_ * H_) / 8, 256, 0, stream>>>(pprm, stats, r_f, k_bar, v_vec, g_vec,
                                                      rho, gam, bet, preout);

    // final projection straight into d_out
    gemm_bf16<0, true, false><<<grid_for(D_), 128, 0, stream>>>(preout, wb_o, nullptr, out_main, nullptr, D_, D_);

    tail_copy<<<(B_ * D_ + 255) / 256, 256, 0, stream>>>(x, out_xlast);
}
